// Agent_22497038697211
// MI455X (gfx1250) — matrix operand order repaired and checked by reading, not compiled, not run on hardware
//
#include <hip/hip_runtime.h>
#include <hip/hip_bf16.h>

// ---------------------------------------------------------------------------
// Types for CDNA5 WMMA (gfx1250, wave32)
// ---------------------------------------------------------------------------
typedef __attribute__((ext_vector_type(16))) __bf16 bf16x16;
typedef __attribute__((ext_vector_type(8)))  float  f32x8;
typedef unsigned short bf16raw;
typedef int i32x4v __attribute__((__vector_size__(16)));

union FragU {
    bf16x16 v;
    unsigned u[8];
};

// Async global->LDS copy (CDNA5). Guarded: host pass / older toolchains fall
// back to direct global loads.
#if defined(__has_builtin)
#if __has_builtin(__builtin_amdgcn_global_load_async_to_lds_b128) && \
    __has_builtin(__builtin_amdgcn_s_wait_asynccnt)
#define HAVE_ASYNC_LDS 1
#endif
#endif

#ifdef HAVE_ASYNC_LDS
// Signature (verified via compiler diagnostics):
//   void __builtin_amdgcn_global_load_async_to_lds_b128(
//       int4 addrspace(1)* src, int4 addrspace(3)* dst, imm int offset, imm int cpol)
__device__ __forceinline__ __attribute__((address_space(1))) i32x4v*
as_global_v4(const void* p) {
    return (__attribute__((address_space(1))) i32x4v*)(unsigned long long)p;
}
__device__ __forceinline__ __attribute__((address_space(3))) i32x4v*
as_lds_v4(void* p) {
    // flat LDS pointer: low 32 bits are the LDS (AS3) address
    return (__attribute__((address_space(3))) i32x4v*)(unsigned)(unsigned long long)p;
}
#endif

__device__ __forceinline__ unsigned short f2bf(float f) {
    unsigned x = __float_as_uint(f);
    x += 0x7FFFu + ((x >> 16) & 1u);   // round-to-nearest-even
    return (unsigned short)(x >> 16);
}
__device__ __forceinline__ unsigned pack2(float lo, float hi) {
    return (unsigned)f2bf(lo) | ((unsigned)f2bf(hi) << 16);
}

// A-fragment (16x32 bf16, MxK) from a bf16 row-major row: two b128 loads.
// lane<16 : row m, K {k0..k0+7} u[0..3], {k0+16..23} u[4..7]
// lane>=16: row m, K {k0+8..15} u[0..3], {k0+24..31} u[4..7]
// maskAnd: 0xFFFFFFFF to keep, 0 to zero (binary done-mask on h).
__device__ __forceinline__ bf16x16 load_afrag(const bf16raw* __restrict__ rowp,
                                              int k0, int hi, unsigned maskAnd) {
    const uint4* p0 = (const uint4*)(rowp + k0 + hi * 8);
    const uint4* p1 = (const uint4*)(rowp + k0 + 16 + hi * 8);
    uint4 a = *p0, b = *p1;
    FragU f;
    f.u[0] = a.x & maskAnd; f.u[1] = a.y & maskAnd;
    f.u[2] = a.z & maskAnd; f.u[3] = a.w & maskAnd;
    f.u[4] = b.x & maskAnd; f.u[5] = b.y & maskAnd;
    f.u[6] = b.z & maskAnd; f.u[7] = b.w & maskAnd;
    return f.v;
}

// B-fragment directly from packed global weights (fallback path).
__device__ __forceinline__ bf16x16 load_bfrag(const unsigned* __restrict__ packed,
                                              size_t fragIdx, int lane) {
    const uint4* p = (const uint4*)(packed + fragIdx * 256 + (size_t)lane * 8);
    uint4 q0 = p[0], q1 = p[1];
    FragU f;
    f.u[0] = q0.x; f.u[1] = q0.y; f.u[2] = q0.z; f.u[3] = q0.w;
    f.u[4] = q1.x; f.u[5] = q1.y; f.u[6] = q1.z; f.u[7] = q1.w;
    return f.v;
}

// B-fragment from an LDS staging buffer (ds_load_b128 x2).
__device__ __forceinline__ bf16x16 lds_bfrag(const unsigned* st, int lane) {
    const uint4* p = (const uint4*)(st + (size_t)lane * 8);
    uint4 q0 = p[0], q1 = p[1];
    FragU f;
    f.u[0] = q0.x; f.u[1] = q0.y; f.u[2] = q0.z; f.u[3] = q0.w;
    f.u[4] = q1.x; f.u[5] = q1.y; f.u[6] = q1.z; f.u[7] = q1.w;
    return f.v;
}

__device__ __forceinline__ float sigmoidf(float x) { return 1.f / (1.f + expf(-x)); }

// ---------------------------------------------------------------------------
// Weight packing: W (N x Kreal, row-major f32) -> bf16 B-fragments.
// Fragment (ntile, kc): 32 lanes x 8 dwords (32B/lane, coalesced b128 loads).
// ---------------------------------------------------------------------------
__global__ __launch_bounds__(256)
void pack_weight_kernel(const float* __restrict__ W, unsigned* __restrict__ out,
                        int N, int Kreal, int Kpad) {
    int KC   = Kpad >> 5;
    int frag = blockIdx.x;
    int kc   = frag % KC;
    int t    = threadIdx.x;
    int lane = t >> 3;
    int d    = t & 7;
    int ntile = frag / KC;
    int n  = ntile * 16 + (lane & 15);
    int kb = kc * 32 + ((d >> 2) << 4) + ((lane >> 4) << 3) + 2 * (d & 3);
    float lo = 0.f, hi = 0.f;
    if (n < N) {
        if (kb     < Kreal) lo = W[(size_t)n * Kreal + kb];
        if (kb + 1 < Kreal) hi = W[(size_t)n * Kreal + kb + 1];
    }
    out[(size_t)frag * 256 + t] = pack2(lo, hi);
}

// ---------------------------------------------------------------------------
// Fused 4-layer CNN + LayerNorm + ReLU + spatial mean, one block per image.
// All intermediates LDS-resident (CDNA5: 320KB/WGP). grid=(2048, 2 cams).
// Emits features directly as bf16 (GEMM A input format).
// ---------------------------------------------------------------------------
struct ConvParams { const float* W[4]; const float* b[4]; const float* g[4]; const float* be[4]; };

__device__ void ln_relu(float* buf, int n, const float* __restrict__ g,
                        const float* __restrict__ be, float* red) {
    int tid = threadIdx.x;
    float s = 0.f, s2 = 0.f;
    for (int i = tid; i < n; i += 256) { float v = buf[i]; s += v; s2 += v * v; }
    red[tid] = s; red[256 + tid] = s2;
    __syncthreads();
    for (int off = 128; off; off >>= 1) {
        if (tid < off) { red[tid] += red[tid + off]; red[256 + tid] += red[256 + tid + off]; }
        __syncthreads();
    }
    float mu = red[0] / (float)n;
    float var = red[256] / (float)n - mu * mu;
    float rs = rsqrtf(var + 1e-5f);
    for (int i = tid; i < n; i += 256) {
        float v = (buf[i] - mu) * rs * g[i] + be[i];
        buf[i] = fmaxf(v, 0.f);
    }
    __syncthreads();
}

template <int CIN, int COUT, int KK, int S, int IH, int IW, int OH, int OW>
__device__ void conv_lds(const float* __restrict__ src, const float* __restrict__ W,
                         const float* __restrict__ b, float* dst) {
    const int NOUT = COUT * OH * OW;
    for (int idx = threadIdx.x; idx < NOUT; idx += 256) {
        int c = idx / (OH * OW);
        int r = idx % (OH * OW);
        int oy = r / OW, ox = r % OW;
        float acc = b[c];
        for (int ci = 0; ci < CIN; ++ci) {
            const float* sp = src + ci * IH * IW + oy * S * IW + ox * S;
            const float* wp = W + ((size_t)(c * CIN + ci)) * KK * KK;
#pragma unroll
            for (int ky = 0; ky < KK; ++ky)
#pragma unroll
                for (int kx = 0; kx < KK; ++kx)
                    acc += sp[ky * IW + kx] * wp[ky * KK + kx];
        }
        dst[idx] = acc;
    }
    __syncthreads();
}

__global__ __launch_bounds__(256)
void conv_tower_kernel(const float* __restrict__ x, ConvParams ph, ConvParams pw,
                       bf16raw* __restrict__ feat) {
    __shared__ float s_l1[16 * 39 * 29];   // 18096
    __shared__ float s_l2[32 * 17 * 12];   // 6528
    __shared__ float s_l3[64 * 7 * 5];     // 2240
    __shared__ float s_l4[128 * 2 * 1];    // 256
    __shared__ float s_red[512];

    int n   = blockIdx.x;
    int cam = blockIdx.y;
    const ConvParams& P = (cam == 0) ? ph : pw;
    const float* img = x + (size_t)n * 38448 + 48 + (size_t)cam * 19200;

    // Layer 1: 1->16, k=8, s=4, src in global (scaled by 1/2)
    for (int idx = threadIdx.x; idx < 16 * 39 * 29; idx += 256) {
        int c = idx / (39 * 29);
        int r = idx % (39 * 29);
        int oy = r / 29, ox = r % 29;
        float acc = P.b[0][c];
        const float* Wc = P.W[0] + (size_t)c * 64;
#pragma unroll
        for (int ky = 0; ky < 8; ++ky)
#pragma unroll
            for (int kx = 0; kx < 8; ++kx)
                acc += 0.5f * img[(oy * 4 + ky) * 120 + (ox * 4 + kx)] * Wc[ky * 8 + kx];
        s_l1[idx] = acc;
    }
    __syncthreads();
    ln_relu(s_l1, 18096, P.g[0], P.be[0], s_red);

    conv_lds<16, 32, 6, 2, 39, 29, 17, 12>(s_l1, P.W[1], P.b[1], s_l2);
    ln_relu(s_l2, 6528, P.g[1], P.be[1], s_red);

    conv_lds<32, 64, 4, 2, 17, 12, 7, 5>(s_l2, P.W[2], P.b[2], s_l3);
    ln_relu(s_l3, 2240, P.g[2], P.be[2], s_red);

    conv_lds<64, 128, 4, 2, 7, 5, 2, 1>(s_l3, P.W[3], P.b[3], s_l4);
    ln_relu(s_l4, 256, P.g[3], P.be[3], s_red);

    if (threadIdx.x < 128)
        feat[(size_t)n * 320 + cam * 128 + threadIdx.x] =
            f2bf(0.5f * (s_l4[threadIdx.x * 2] + s_l4[threadIdx.x * 2 + 1]));
}

// feat[:, 256:304] = proprio (bf16), feat[:, 304:320] = 0 (K padding for WMMA)
__global__ __launch_bounds__(256)
void proprio_kernel(const float* __restrict__ x, bf16raw* __restrict__ feat) {
    int i = blockIdx.x * 256 + threadIdx.x;
    if (i < 2048 * 64) {
        int row = i >> 6, j = i & 63;
        feat[(size_t)row * 320 + 256 + j] =
            (j < 48) ? f2bf(x[(size_t)row * 38448 + j]) : (bf16raw)0;
    }
}

__global__ __launch_bounds__(256)
void init_state_kernel(const float* __restrict__ h0, const float* __restrict__ c0,
                       bf16raw* hA, bf16raw* hB, float* cA, float* cB) {
    int i = blockIdx.x * 256 + threadIdx.x;
    if (i < 128 * 1024) {
        hA[i] = f2bf(h0[i]);
        hB[i] = f2bf(h0[131072 + i]);
        cA[i] = c0[i];
        cB[i] = c0[131072 + i];
    }
}

// ---------------------------------------------------------------------------
// LSTM cell: block(128)=4 waves, wave w owns gate w (i,f,g,o); 16x64 h/c tile.
// grid = (8, 16). B fragments double-buffered in LDS via async global->LDS:
// issue(kc+1) -> s_wait_asynccnt -> 8x ds_load_b128 -> 4x v_wmma back-to-back.
// ---------------------------------------------------------------------------
__global__ __launch_bounds__(128)
void lstm_cell_kernel(const bf16raw* __restrict__ X, int ldx, int KCx,
                      const unsigned* __restrict__ Wihp, const unsigned* __restrict__ Whhp,
                      const float* __restrict__ bih, const float* __restrict__ bhh,
                      const bf16raw* __restrict__ h_in, bf16raw* __restrict__ h_out,
                      float* __restrict__ c_io, const int* __restrict__ done_t,
                      bf16raw* __restrict__ outp) {
    __shared__ float sg[4][16][64];
#ifdef HAVE_ASYNC_LDS
    __shared__ unsigned stage[4][2][4][256];   // [wave][buf][ct][dwords]
#endif

    int gate = threadIdx.x >> 5;
    int lane = threadIdx.x & 31;
    int hi = lane >> 4;
    int mr = lane & 15;
    int mtile = blockIdx.x, by = blockIdx.y;
    int row = mtile * 16 + mr;
    unsigned hmask = done_t[row] ? 0u : 0xFFFFFFFFu;   // binary done-mask on h

    f32x8 acc[4];
#pragma unroll
    for (int ct = 0; ct < 4; ++ct)
#pragma unroll
        for (int r = 0; r < 8; ++r) acc[ct][r] = 0.f;

    const int ntb = gate * 64 + by * 4;   // first of this wave's 4 ntiles

    // Phase 1: X * Wih^T          Phase 2: (mask.h_prev) * Whh^T
    const bf16raw* xr = X + (size_t)row * ldx;
    const bf16raw* hr = h_in + (size_t)row * 1024;
    for (int phase = 0; phase < 2; ++phase) {
        const unsigned* Wp = phase ? Whhp : Wihp;
        const bf16raw* ar  = phase ? hr : xr;
        unsigned amask     = phase ? hmask : 0xFFFFFFFFu;
        int KC             = phase ? 32 : KCx;
#ifdef HAVE_ASYNC_LDS
        // Loop-invariant per-ct byte bases (strength-reduced: +1024B per kc).
        const char* srcb[4];
#pragma unroll
        for (int ct = 0; ct < 4; ++ct)
            srcb[ct] = (const char*)Wp + ((size_t)(ntb + ct) * KC) * 1024 + (size_t)lane * 32;
        auto issue = [&](int kc, int buf) {
            size_t koff = (size_t)kc * 1024;
#pragma unroll
            for (int ct = 0; ct < 4; ++ct) {
                const char* src = srcb[ct] + koff;
                unsigned* dst = &stage[gate][buf][ct][lane * 8];
                __builtin_amdgcn_global_load_async_to_lds_b128(as_global_v4(src), as_lds_v4(dst), 0, 0);
                __builtin_amdgcn_global_load_async_to_lds_b128(as_global_v4(src), as_lds_v4(dst), 16, 0);
            }
        };
        int buf = 0;
        issue(0, 0);
        for (int kc = 0; kc < KC; ++kc) {
            if (kc + 1 < KC) {
                issue(kc + 1, buf ^ 1);
                __builtin_amdgcn_s_wait_asynccnt(8);
            } else {
                __builtin_amdgcn_s_wait_asynccnt(0);
            }
            bf16x16 a = load_afrag(ar, kc * 32, hi, amask);
            // Load all 4 B fragments first (8 independent ds_load_b128),
            // then chain the 4 WMMAs back-to-back.
            bf16x16 b0 = lds_bfrag(&stage[gate][buf][0][0], lane);
            bf16x16 b1 = lds_bfrag(&stage[gate][buf][1][0], lane);
            bf16x16 b2 = lds_bfrag(&stage[gate][buf][2][0], lane);
            bf16x16 b3 = lds_bfrag(&stage[gate][buf][3][0], lane);
            acc[0] = __builtin_amdgcn_wmma_f32_16x16x32_bf16(false, a, false, b0, (short)0, acc[0], false, false);
            acc[1] = __builtin_amdgcn_wmma_f32_16x16x32_bf16(false, a, false, b1, (short)0, acc[1], false, false);
            acc[2] = __builtin_amdgcn_wmma_f32_16x16x32_bf16(false, a, false, b2, (short)0, acc[2], false, false);
            acc[3] = __builtin_amdgcn_wmma_f32_16x16x32_bf16(false, a, false, b3, (short)0, acc[3], false, false);
            buf ^= 1;
        }
#else
        for (int kc = 0; kc < KC; ++kc) {
            bf16x16 a = load_afrag(ar, kc * 32, hi, amask);
            bf16x16 b0 = load_bfrag(Wp, (size_t)(ntb + 0) * KC + kc, lane);
            bf16x16 b1 = load_bfrag(Wp, (size_t)(ntb + 1) * KC + kc, lane);
            bf16x16 b2 = load_bfrag(Wp, (size_t)(ntb + 2) * KC + kc, lane);
            bf16x16 b3 = load_bfrag(Wp, (size_t)(ntb + 3) * KC + kc, lane);
            if (kc + 1 < KC)
                __builtin_prefetch(Wp + ((size_t)ntb * KC + kc + 1) * 256 + (size_t)lane * 8, 0, 1);
            acc[0] = __builtin_amdgcn_wmma_f32_16x16x32_bf16(false, a, false, b0, (short)0, acc[0], false, false);
            acc[1] = __builtin_amdgcn_wmma_f32_16x16x32_bf16(false, a, false, b1, (short)0, acc[1], false, false);
            acc[2] = __builtin_amdgcn_wmma_f32_16x16x32_bf16(false, a, false, b2, (short)0, acc[2], false, false);
            acc[3] = __builtin_amdgcn_wmma_f32_16x16x32_bf16(false, a, false, b3, (short)0, acc[3], false, false);
        }
#endif
    }

    // Stash gate tiles in LDS (C layout: lane -> N, vgpr r -> M = r + 8*hi)
#pragma unroll
    for (int ct = 0; ct < 4; ++ct) {
        int nn = by * 64 + ct * 16 + mr;
        int gc = gate * 1024 + nn;
        float bsum = bih[gc] + bhh[gc];
#pragma unroll
        for (int r = 0; r < 8; ++r) sg[gate][r + hi * 8][ct * 16 + mr] = acc[ct][r] + bsum;
    }
    __syncthreads();

    // Fused gate math + state update (c in f32, h emitted as bf16)
    for (int e = threadIdx.x; e < 1024; e += 128) {
        int mrr = e >> 6, nl = e & 63;
        int rw = mtile * 16 + mrr;
        int nn = by * 64 + nl;
        float mm = done_t[rw] ? 0.f : 1.f;
        float i_ = sigmoidf(sg[0][mrr][nl]);
        float f_ = sigmoidf(sg[1][mrr][nl]);
        float g_ = tanhf(sg[2][mrr][nl]);
        float o_ = sigmoidf(sg[3][mrr][nl]);
        float cold = c_io[(size_t)rw * 1024 + nn] * mm;
        float cn = f_ * cold + i_ * g_;
        float hn = o_ * tanhf(cn);
        c_io[(size_t)rw * 1024 + nn] = cn;
        bf16raw hb = f2bf(hn);
        h_out[(size_t)rw * 1024 + nn] = hb;
        if (outp) outp[(size_t)rw * 1024 + nn] = hb;
    }
}

// ---------------------------------------------------------------------------
// Generic GEMM: out = act(X * W^T + b). X bf16 row-major, W pre-packed bf16.
// block(128)=4 waves, wave w -> col-tile w. grid = (M/16, N/64).
// ---------------------------------------------------------------------------
__global__ __launch_bounds__(128)
void gemm_bias_act_kernel(const bf16raw* __restrict__ X, int ldx, int KC,
                          const unsigned* __restrict__ Wp, const float* __restrict__ bias,
                          int N, void* __restrict__ out, int elu, int out_bf16) {
#ifdef HAVE_ASYNC_LDS
    __shared__ unsigned stage[4][2][256];
#endif
    int w = threadIdx.x >> 5;
    int lane = threadIdx.x & 31;
    int hi = lane >> 4;
    int mr = lane & 15;
    int mtile = blockIdx.x;
    int ntile = blockIdx.y * 4 + w;
    const bf16raw* rowp = X + (size_t)(mtile * 16 + mr) * ldx;

    f32x8 acc;
#pragma unroll
    for (int r = 0; r < 8; ++r) acc[r] = 0.f;

#ifdef HAVE_ASYNC_LDS
    const char* srcb = (const char*)Wp + ((size_t)ntile * KC) * 1024 + (size_t)lane * 32;
    auto issue = [&](int kc, int buf) {
        const char* src = srcb + (size_t)kc * 1024;
        unsigned* dst = &stage[w][buf][lane * 8];
        __builtin_amdgcn_global_load_async_to_lds_b128(as_global_v4(src), as_lds_v4(dst), 0, 0);
        __builtin_amdgcn_global_load_async_to_lds_b128(as_global_v4(src), as_lds_v4(dst), 16, 0);
    };
    int buf = 0;
    issue(0, 0);
    for (int kc = 0; kc < KC; ++kc) {
        if (kc + 1 < KC) {
            issue(kc + 1, buf ^ 1);
            __builtin_amdgcn_s_wait_asynccnt(2);
        } else {
            __builtin_amdgcn_s_wait_asynccnt(0);
        }
        bf16x16 a = load_afrag(rowp, kc * 32, hi, 0xFFFFFFFFu);
        bf16x16 b = lds_bfrag(&stage[w][buf][0], lane);
        acc = __builtin_amdgcn_wmma_f32_16x16x32_bf16(
            false, a, false, b, (short)0, acc, false, false);
        buf ^= 1;
    }
#else
    for (int kc = 0; kc < KC; ++kc) {
        if (kc + 1 < KC)
            __builtin_prefetch(Wp + ((size_t)ntile * KC + kc + 1) * 256 + (size_t)lane * 8, 0, 1);
        bf16x16 a = load_afrag(rowp, kc * 32, hi, 0xFFFFFFFFu);
        bf16x16 b = load_bfrag(Wp, (size_t)ntile * KC + kc, lane);
        acc = __builtin_amdgcn_wmma_f32_16x16x32_bf16(
            false, a, false, b, (short)0, acc, false, false);
    }
#endif

    int nn = ntile * 16 + mr;
    float bv = bias[nn];
#pragma unroll
    for (int r = 0; r < 8; ++r) {
        int mm = mtile * 16 + r + hi * 8;
        float v = acc[r] + bv;
        if (elu) v = (v > 0.f) ? v : expm1f(v);
        if (out_bf16) ((bf16raw*)out)[(size_t)mm * N + nn] = f2bf(v);
        else          ((float*)out)[(size_t)mm * N + nn] = v;
    }
}

// ---------------------------------------------------------------------------
// Heads: out[row,0:7] = h @ am_W^T + am_b ; out[row,7] = h @ cr_W^T + cr_b
// 8 heads x 32 lanes per row; wave32 shuffle reduction.
// ---------------------------------------------------------------------------
__global__ __launch_bounds__(256)
void heads_kernel(const float* __restrict__ h, const float* __restrict__ amW,
                  const float* __restrict__ amb, const float* __restrict__ crW,
                  const float* __restrict__ crb, float* __restrict__ out) {
    int row = blockIdx.x;
    int head = threadIdx.x >> 5;
    int lane = threadIdx.x & 31;
    const float* wv = (head < 7) ? amW + (size_t)head * 256 : crW;
    float bias = (head < 7) ? amb[head] : crb[0];
    const float* hr = h + (size_t)row * 256;
    float s = 0.f;
    for (int j = lane; j < 256; j += 32) s += hr[j] * wv[j];
#pragma unroll
    for (int o = 16; o; o >>= 1) s += __shfl_down(s, o);
    if (lane == 0) out[(size_t)row * 8 + head] = s + bias;
}

// ---------------------------------------------------------------------------
// Host launcher
// ---------------------------------------------------------------------------
extern "C" void kernel_launch(void* const* d_in, const int* in_sizes, int n_in,
                              void* d_out, int out_size, void* d_ws, size_t ws_size,
                              hipStream_t stream) {
    (void)in_sizes; (void)n_in; (void)out_size; (void)ws_size;

    const float* x    = (const float*)d_in[0];
    const int*   done = (const int*)d_in[1];
    const float* h0   = (const float*)d_in[2];
    const float* c0   = (const float*)d_in[3];

    ConvParams ph, pw;
    for (int l = 0; l < 4; ++l) {
        ph.W[l]  = (const float*)d_in[4 + l * 4];
        ph.b[l]  = (const float*)d_in[5 + l * 4];
        ph.g[l]  = (const float*)d_in[6 + l * 4];
        ph.be[l] = (const float*)d_in[7 + l * 4];
        pw.W[l]  = (const float*)d_in[20 + l * 4];
        pw.b[l]  = (const float*)d_in[21 + l * 4];
        pw.g[l]  = (const float*)d_in[22 + l * 4];
        pw.be[l] = (const float*)d_in[23 + l * 4];
    }
    const float* Wih1 = (const float*)d_in[36];
    const float* Whh1 = (const float*)d_in[37];
    const float* bih1 = (const float*)d_in[38];
    const float* bhh1 = (const float*)d_in[39];
    const float* Wih2 = (const float*)d_in[40];
    const float* Whh2 = (const float*)d_in[41];
    const float* bih2 = (const float*)d_in[42];
    const float* bhh2 = (const float*)d_in[43];
    const float* mW1 = (const float*)d_in[44];
    const float* mb1 = (const float*)d_in[45];
    const float* mW2 = (const float*)d_in[46];
    const float* mb2 = (const float*)d_in[47];
    const float* mW3 = (const float*)d_in[48];
    const float* mb3 = (const float*)d_in[49];
    const float* amW = (const float*)d_in[50];
    const float* amb = (const float*)d_in[51];
    const float* crW = (const float*)d_in[52];
    const float* crb = (const float*)d_in[53];

    // Workspace carve-up
    char* p = (char*)d_ws;
    auto alloc = [&](size_t bytes) -> void* {
        void* r = (void*)p;
        p += (bytes + 255) & ~(size_t)255;
        return r;
    };
    bf16raw* feat  = (bf16raw*)alloc((size_t)2048 * 320 * 2);
    bf16raw* hA[2] = { (bf16raw*)alloc(131072 * 2), (bf16raw*)alloc(131072 * 2) };
    bf16raw* hB[2] = { (bf16raw*)alloc(131072 * 2), (bf16raw*)alloc(131072 * 2) };
    float* cA      = (float*)alloc(131072 * 4);
    float* cB      = (float*)alloc(131072 * 4);
    bf16raw* outs  = (bf16raw*)alloc((size_t)2048 * 1024 * 2);
    bf16raw* m1    = (bf16raw*)alloc((size_t)2048 * 512 * 2);
    bf16raw* m2    = (bf16raw*)alloc((size_t)2048 * 512 * 2);
    float* m3      = (float*)alloc((size_t)2048 * 256 * 4);
    unsigned* p_ih1 = (unsigned*)alloc((size_t)256 * 10 * 256 * 4);
    unsigned* p_hh1 = (unsigned*)alloc((size_t)256 * 32 * 256 * 4);
    unsigned* p_ih2 = (unsigned*)alloc((size_t)256 * 32 * 256 * 4);
    unsigned* p_hh2 = (unsigned*)alloc((size_t)256 * 32 * 256 * 4);
    unsigned* pm1   = (unsigned*)alloc((size_t)32 * 32 * 256 * 4);
    unsigned* pm2   = (unsigned*)alloc((size_t)32 * 16 * 256 * 4);
    unsigned* pm3   = (unsigned*)alloc((size_t)16 * 16 * 256 * 4);

    // 1) Pack GEMM weights to bf16 B-fragment layout (L2-resident afterwards)
    pack_weight_kernel<<<256 * 10, 256, 0, stream>>>(Wih1, p_ih1, 4096, 304, 320);
    pack_weight_kernel<<<256 * 32, 256, 0, stream>>>(Whh1, p_hh1, 4096, 1024, 1024);
    pack_weight_kernel<<<256 * 32, 256, 0, stream>>>(Wih2, p_ih2, 4096, 1024, 1024);
    pack_weight_kernel<<<256 * 32, 256, 0, stream>>>(Whh2, p_hh2, 4096, 1024, 1024);
    pack_weight_kernel<<<32 * 32, 256, 0, stream>>>(mW1, pm1, 512, 1024, 1024);
    pack_weight_kernel<<<32 * 16, 256, 0, stream>>>(mW2, pm2, 512, 512, 512);
    pack_weight_kernel<<<16 * 16, 256, 0, stream>>>(mW3, pm3, 256, 512, 512);

    // 2) Features: proprio + zero-pad, then fused CNN towers (LDS-resident)
    proprio_kernel<<<512, 256, 0, stream>>>(x, feat);
    conv_tower_kernel<<<dim3(2048, 2), 256, 0, stream>>>(x, ph, pw, feat);

    // 3) LSTM init + 16 sequential steps x 2 cells
    init_state_kernel<<<512, 256, 0, stream>>>(h0, c0, hA[0], hB[0], cA, cB);
    for (int t = 0; t < 16; ++t) {
        int cur = t & 1, nxt = cur ^ 1;
        lstm_cell_kernel<<<dim3(8, 16), 128, 0, stream>>>(
            feat + (size_t)t * 128 * 320, 320, 10, p_ih1, p_hh1, bih1, bhh1,
            hA[cur], hA[nxt], cA, done + t * 128, (bf16raw*)nullptr);
        lstm_cell_kernel<<<dim3(8, 16), 128, 0, stream>>>(
            hA[nxt], 1024, 32, p_ih2, p_hh2, bih2, bhh2,
            hB[cur], hB[nxt], cB, done + t * 128, outs + (size_t)t * 128 * 1024);
    }

    // 4) MLP (ELU) + heads
    gemm_bias_act_kernel<<<dim3(128, 8), 128, 0, stream>>>(outs, 1024, 32, pm1, mb1, 512, m1, 1, 1);
    gemm_bias_act_kernel<<<dim3(128, 8), 128, 0, stream>>>(m1, 512, 16, pm2, mb2, 512, m2, 1, 1);
    gemm_bias_act_kernel<<<dim3(128, 4), 128, 0, stream>>>(m2, 512, 16, pm3, mb3, 256, m3, 1, 0);
    heads_kernel<<<2048, 256, 0, stream>>>(m3, amW, amb, crW, crb, (float*)d_out);
}
